// Convolution_33809982554737
// MI455X (gfx1250) — matrix operand order repaired and checked by reading, not compiled, not run on hardware
//
#include <hip/hip_runtime.h>
#include <hip/hip_bf16.h>
#include <math.h>

typedef __attribute__((ext_vector_type(16))) _Float16 v16h;
typedef __attribute__((ext_vector_type(8)))  _Float16 v8h;
typedef __attribute__((ext_vector_type(8)))  float    v8f;
typedef __attribute__((ext_vector_type(4)))  unsigned int v4u;
typedef __attribute__((ext_vector_type(8)))  int      v8i;
typedef __attribute__((ext_vector_type(4)))  int      v4i;

#define TAPS 27
#define CIN  32
#define COUT 64
#define KBYTES (TAPS * 4 * 32 * 16 * 2)   // 110592 bytes of f16 kernel matrix

// ---------------------------------------------------------------------------
// Kernel 1: build fused conv kernel matrix K[27][32][64] (f16) directly in
// WMMA B-fragment layout: idx = ((tap*4 + ntile)*32 + lane)*16 + e, where
// lane = (ci>=16 ? 16:0) + (co&15), e = ci&15, ntile = co>>4.
// Self-connection folded into center tap (13).
// ---------------------------------------------------------------------------
__device__ __forceinline__ float rad4(const float* __restrict__ w,
                                      const float emb[4], int u, int wcol) {
    float s = 0.f;
#pragma unroll
    for (int k = 0; k < 4; ++k) s += emb[k] * w[k * 128 + u * 16 + wcol];
    return s * (1.0f / 27.0f);
}

__global__ __launch_bounds__(256) void build_kmat(
    const float* __restrict__ w_lin0, const float* __restrict__ w_lin1,
    const float* __restrict__ w000,   const float* __restrict__ w011,
    const float* __restrict__ w101,   const float* __restrict__ w110,
    _Float16* __restrict__ kb)
{
    int gid = blockIdx.x * 256 + threadIdx.x;
    if (gid >= TAPS * CIN * COUT) return;
    int t   = gid >> 11;
    int rem = gid & 2047;
    int ci  = rem >> 6;
    int co  = rem & 63;

    int dx = t / 9, dy = (t / 3) % 3, dz = t % 3;
    float lx = (float)(dx - 1), ly = (float)(dy - 1), lz = (float)(dz - 1);
    float dist = sqrtf(lx * lx + ly * ly + lz * lz);

    float emb[4];
    const float scale = 1.14136f * expf(2.0f);
#pragma unroll
    for (int k = 0; k < 4; ++k) {
        float diff = (dist - 0.3f * (float)(k + 1)) * (1.0f / 0.3f);
        float a = diff + 1.0f, b = 1.0f - diff;
        float sa = (a > 0.0f) ? expf(-1.0f / a) : 0.0f;
        float sb = (b > 0.0f) ? expf(-1.0f / b) : 0.0f;
        emb[k] = scale * sa * sb;
    }
    float inv = (dist > 0.0f) ? (1.0f / dist) : 0.0f;
    float sh1[3];
    sh1[0] = 1.7320508075688772f * lx * inv;
    sh1[1] = 1.7320508075688772f * ly * inv;
    sh1[2] = 1.7320508075688772f * lz * inv;

    const float alpha  = 0.25f;
    const float inv_s3 = 0.5773502691896258f;

    float val;
    if (ci < 8) {
        int u = ci;
        if (co < 16) {
            val = alpha * rad4(w000, emb, u, co);
        } else {
            int cc = co - 16, wcol = cc / 3, m = cc % 3;
            val = alpha * inv_s3 * rad4(w011, emb, u, wcol) * sh1[m];
        }
    } else {
        int u = (ci - 8) / 3, i = (ci - 8) % 3;
        if (co < 16) {
            val = alpha * inv_s3 * rad4(w110, emb, u, co) * sh1[i];
        } else {
            int cc = co - 16, wcol = cc / 3, m = cc % 3;
            val = (i == m) ? alpha * inv_s3 * rad4(w101, emb, u, wcol) : 0.0f;
        }
    }
    if (t == 13) {
        const float lin_norm = 0.35355339059327373f;
        if (ci < 8 && co < 16) {
            val += lin_norm * w_lin0[ci * 16 + co];
        } else if (ci >= 8 && co >= 16) {
            int u = (ci - 8) / 3, i = (ci - 8) % 3;
            int cc = co - 16, wcol = cc / 3, m = cc % 3;
            if (i == m) val += lin_norm * w_lin1[u * 16 + wcol];
        }
    }
    int nt = co >> 4, n = co & 15, khalf = ci >> 4, e = ci & 15;
    kb[((t * 4 + nt) * 32 + (khalf * 16 + n)) * 16 + e] = (_Float16)val;
}

// ---------------------------------------------------------------------------
// Kernel 2: implicit-GEMM conv. Workgroup = (batch, x-plane, y-half(16 rows)).
// 8 waves; each wave owns 2 y rows x 2 z-tiles = 4 M-tiles (128 acc VGPRs),
// so every B fragment read from LDS feeds 4 WMMAs. Kernel matrix brought into
// LDS by the Tensor Data Mover (TENSORcnt), x staged f32->f16 with halos.
// ---------------------------------------------------------------------------
__global__ __launch_bounds__(256) void conv_wmma(
    const float* __restrict__ x, const _Float16* __restrict__ kb,
    float* __restrict__ out)
{
    // Xs[xplane 0..2][ylocal 0..17][zpad 0..33][c 0..31]
    __shared__ __align__(64) _Float16 Xs[3 * 18 * 34 * 32];   // 117504 B
    __shared__ __align__(64) _Float16 Ks[TAPS * 4 * 32 * 16]; // 110592 B

    const int tid = threadIdx.x;
    const int wg  = blockIdx.x;            // 0..511
    const int y0  = (wg & 1) * 16;
    const int xx  = (wg >> 1) & 31;
    const int b   = wg >> 6;

    // ---- TDM: pull the 110592-byte kernel matrix into LDS (wave 0) ----
    if (tid < 32) {
        unsigned long long ga = (unsigned long long)(uintptr_t)kb;
        unsigned int lds = (unsigned int)(uintptr_t)(&Ks[0]);
        const unsigned int NU = KBYTES / 8;          // 13824 units of 8 bytes
        v4u g0;
        g0.x = 1u;                                   // count=1, valid user D#
        g0.y = lds;                                  // lds_addr
        g0.z = (unsigned int)ga;                     // global_addr[31:0]
        g0.w = (unsigned int)((ga >> 32) & 0x01FFFFFFu) | (2u << 30); // type=2
        v8i g1;
        g1[0] = (int)(3u << 16);                     // data_size = 8 bytes
        g1[1] = (int)((NU & 0xFFFFu) << 16);         // tensor_dim0[15:0]
        g1[2] = (int)((NU >> 16) | (1u << 16));      // dim0[31:16], tensor_dim1=1
        g1[3] = (int)((NU & 0xFFFFu) << 16);         // tile_dim0 = NU
        g1[4] = 1;                                   // tile_dim1=1, tile_dim2=0
        g1[5] = (int)NU;                             // tensor_dim0_stride[31:0]
        g1[6] = 0;
        g1[7] = 0;
        v4i gz4 = (v4i)0;
        v8i gz8 = (v8i)0;
        __builtin_amdgcn_tensor_load_to_lds(g0, g1, gz4, gz4, gz8, 0);
        __builtin_amdgcn_s_wait_tensorcnt(0);
    }

    // ---- stage x (f32 -> f16), zeroing out-of-grid x/y halos ----
    for (int i = tid; i < 3 * 18 * 32 * 8; i += 256) {
        int c4 = i & 7;
        int z  = (i >> 3) & 31;
        int t2 = i >> 8;              // 0..53
        int yl = t2 % 18;
        int xp = t2 / 18;
        int gx = xx + xp - 1;
        int gy = y0 + yl - 1;
        float4 v = make_float4(0.f, 0.f, 0.f, 0.f);
        if ((unsigned)gx < 32u && (unsigned)gy < 32u) {
            const float* p = x + (((((size_t)b * 32 + gx) * 32 + gy) * 32 + z) * 32 + c4 * 4);
            v = *(const float4*)p;
        }
        _Float16* d = &Xs[((xp * 18 + yl) * 34 + z + 1) * 32 + c4 * 4];
        d[0] = (_Float16)v.x; d[1] = (_Float16)v.y;
        d[2] = (_Float16)v.z; d[3] = (_Float16)v.w;
    }
    for (int i = tid; i < 3 * 18 * 2 * 8; i += 256) {   // zero z pads
        int c4 = i & 7;
        int zp = (i >> 3) & 1;
        int t2 = i >> 4;
        int yl = t2 % 18;
        int xp = t2 / 18;
        _Float16* d = &Xs[((xp * 18 + yl) * 34 + zp * 33) * 32 + c4 * 4];
        d[0] = (_Float16)0.f; d[1] = (_Float16)0.f;
        d[2] = (_Float16)0.f; d[3] = (_Float16)0.f;
    }
    __syncthreads();

    const int lane  = tid & 31;
    const int wid   = tid >> 5;        // 0..7
    const int row0  = wid * 2;         // local y row base (0..14)
    const int mrow  = lane & 15;       // A row M (z in tile) / C col N
    const int khalf = lane >> 4;

    v8f acc[2][2][4];                  // [yrow][ztile][ntile]
#pragma unroll
    for (int r = 0; r < 2; ++r)
#pragma unroll
        for (int zt = 0; zt < 2; ++zt)
#pragma unroll
            for (int nt = 0; nt < 4; ++nt) acc[r][zt][nt] = (v8f)0.0f;

    for (int t = 0; t < TAPS; ++t) {
        int dx = t / 9, dy = (t / 3) % 3, dz = t % 3;
        // A fragments (16x32 f16): lanes<16 hold K{0..15}, lanes>=16
        // hold K{16..31}; row M = lane&15 = z within tile.
        v16h A[2][2];
#pragma unroll
        for (int r = 0; r < 2; ++r) {
            const _Float16* ab =
                &Xs[((dx * 18 + row0 + r + dy) * 34 + dz + mrow) * 32];
            v8h c1a = *(const v8h*)(ab + khalf * 16);
            v8h c2a = *(const v8h*)(ab + 8 + khalf * 16);
            A[r][0] = __builtin_shufflevector(c1a, c2a,
                        0,1,2,3,4,5,6,7,8,9,10,11,12,13,14,15);
            v8h c1b = *(const v8h*)(ab + 512 + khalf * 16);   // z tile 1
            v8h c2b = *(const v8h*)(ab + 512 + 8 + khalf * 16);
            A[r][1] = __builtin_shufflevector(c1b, c2b,
                        0,1,2,3,4,5,6,7,8,9,10,11,12,13,14,15);
        }
        const _Float16* kp = &Ks[((t * 4) * 32 + lane) * 16];
#pragma unroll
        for (int nt = 0; nt < 4; ++nt) {
            v16h Bt = *(const v16h*)(kp + nt * 512);
#pragma unroll
            for (int r = 0; r < 2; ++r)
#pragma unroll
                for (int zt = 0; zt < 2; ++zt)
                    acc[r][zt][nt] = __builtin_amdgcn_wmma_f32_16x16x32_f16(
                        false, A[r][zt], false, Bt, (short)0,
                        acc[r][zt][nt], false, false);
        }
    }

    // ---- store: C/D layout: VGPR rr, lanes 0-15 -> M=rr, lanes16-31 -> M=8+rr
#pragma unroll
    for (int r = 0; r < 2; ++r) {
        const int gy = y0 + row0 + r;
        const size_t obase = ((((size_t)b * 32 + xx) * 32 + gy) * 32) * 64;
#pragma unroll
        for (int zt = 0; zt < 2; ++zt)
#pragma unroll
            for (int nt = 0; nt < 4; ++nt)
#pragma unroll
                for (int rr = 0; rr < 8; ++rr) {
                    int z = zt * 16 + khalf * 8 + rr;
                    out[obase + (size_t)z * 64 + nt * 16 + mrow] =
                        acc[r][zt][nt][rr];
                }
    }
}

// ---------------------------------------------------------------------------
extern "C" void kernel_launch(void* const* d_in, const int* in_sizes, int n_in,
                              void* d_out, int out_size, void* d_ws, size_t ws_size,
                              hipStream_t stream) {
    const float* x      = (const float*)d_in[0];
    const float* w_lin0 = (const float*)d_in[1];
    const float* w_lin1 = (const float*)d_in[2];
    const float* w000   = (const float*)d_in[3];
    const float* w011   = (const float*)d_in[4];
    const float* w101   = (const float*)d_in[5];
    const float* w110   = (const float*)d_in[6];
    _Float16* kb = (_Float16*)d_ws;       // 110592 bytes
    float* out = (float*)d_out;

    build_kmat<<<(TAPS * CIN * COUT + 255) / 256, 256, 0, stream>>>(
        w_lin0, w_lin1, w000, w011, w101, w110, kb);
    conv_wmma<<<512, 256, 0, stream>>>(x, kb, out);
}
